// XBOXAttention16_15650860826847
// MI455X (gfx1250) — matrix operand order repaired and checked by reading, compiled, not run on hardware
//
#include <hip/hip_runtime.h>

typedef __attribute__((ext_vector_type(16))) _Float16 v16h;
typedef __attribute__((ext_vector_type(8)))  float    v8f;

#define SLEN 2048
#define DQK  64
#define DA   65

// ---------------------------------------------------------------------------
// Kernel 0: init the phi accumulator (ws is poisoned with 0xAA by harness)
// ---------------------------------------------------------------------------
__global__ void init_phi(unsigned int* phi_bits) {
    *phi_bits = 0u;   // max of non-negative floats via uint bit compare
}

// ---------------------------------------------------------------------------
// Kernel 1: per-row ||qk||^2 and global max (phi^2) via atomicMax on bits
// ---------------------------------------------------------------------------
__global__ __launch_bounds__(256)
void norms_phi(const _Float16* __restrict__ qk,
               float* __restrict__ norm2, unsigned int* __restrict__ phi_bits) {
    const int row = blockIdx.x * blockDim.x + threadIdx.x;   // B*H*S rows
    const _Float16* p = qk + (size_t)row * DQK;
    float s = 0.0f;
#pragma unroll
    for (int d = 0; d < DQK; ++d) { float v = (float)p[d]; s += v * v; }
    norm2[row] = s;
    // wave-reduce max, then one atomic per wave
    float m = s;
#pragma unroll
    for (int off = 16; off > 0; off >>= 1)
        m = fmaxf(m, __shfl_xor(m, off, 32));
    if ((threadIdx.x & 31) == 0)
        atomicMax(phi_bits, __float_as_uint(m));
}

// ---------------------------------------------------------------------------
// Kernel 2: qk_const[i] = sqrt(phi^2 - n2[i]);  Qs[i] = (phi/||qk_i||)*dot(qk_i, a_i[0:64])
// ---------------------------------------------------------------------------
__global__ __launch_bounds__(256)
void qs_qc(const _Float16* __restrict__ qk, const _Float16* __restrict__ a,
           const float* __restrict__ norm2, const unsigned int* __restrict__ phi_bits,
           float* __restrict__ qc_ws, float* __restrict__ qs_ws) {
    const int row = blockIdx.x * blockDim.x + threadIdx.x;
    const float phi2 = __uint_as_float(*phi_bits);           // max ||qk||^2
    const float n2   = norm2[row];
    const float qc   = sqrtf(fmaxf(phi2 - n2, 0.0f));
    const float scale = (n2 > 0.0f) ? sqrtf(phi2) / sqrtf(n2) : 0.0f;
    const _Float16* pq = qk + (size_t)row * DQK;
    const _Float16* pa = a  + (size_t)row * DA;
    float dot = 0.0f;
#pragma unroll
    for (int d = 0; d < DQK; ++d) dot += (float)pq[d] * (float)pa[d];
    qc_ws[row] = qc;
    qs_ws[row] = scale * dot;
}

// ---------------------------------------------------------------------------
// Kernel 3: fused WMMA GEMM strip + rank-1 fixup + scale + top-32 + mask write
//   grid: (B*H) * (S/16) workgroups, 256 threads (8 waves)
//   wave w computes D-tiles for i-columns [chunk*128 + 16w, +16) over all chunks
// ---------------------------------------------------------------------------
__global__ __launch_bounds__(256)
void gemm_topk(const _Float16* __restrict__ qk, const _Float16* __restrict__ a,
               const float* __restrict__ qc_ws, const float* __restrict__ qs_ws,
               float* __restrict__ out) {
    __shared__ _Float16 strip[16 * SLEN];   // 64 KB result strip (fp16, like ref)

    const int bh   = blockIdx.x >> 7;       // S/16 = 128 j-tiles per (b,h)
    const int j0   = (blockIdx.x & 127) * 16;
    const int lane = threadIdx.x & 31;
    const int wave = threadIdx.x >> 5;      // 0..7
    const int n16  = lane & 15;             // N-index / M-row within half-wave
    const int hi   = lane >> 4;             // half-wave select

    const _Float16* qkB = qk + (size_t)bh * SLEN * DQK;
    const _Float16* aB  = a  + (size_t)bh * SLEN * DA;
    const float*    qcB = qc_ws + (size_t)bh * SLEN;
    const float*    qsB = qs_ws + (size_t)bh * SLEN;

    // ---- A fragments: a-tile rows j0..j0+15, K = d (16-bit A 16x32 layout) ----
    // lane<16: row M=lane, K-blocks {0..7, 16..23}; lane>=16: row M=lane-16, {8..15, 24..31}
    v16h A0, A1;
    {
        const _Float16* arow = aB + (size_t)(j0 + n16) * DA;
        const int k0 = hi ? 8 : 0;
#pragma unroll
        for (int e = 0; e < 8; ++e) {
            A0[e]     = arow[k0 + e];            // K = k0+e
            A0[8 + e] = arow[k0 + 16 + e];       // K = k0+16+e
            A1[e]     = arow[32 + k0 + e];       // K = 32+k0+e
            A1[8 + e] = arow[32 + k0 + 16 + e];  // K = 48+k0+e (i.e. 32+k0+16+e)
        }
    }
    // a's 65th column for the 8 j-rows this lane touches in the D fragment
    float alast[8];
#pragma unroll
    for (int r = 0; r < 8; ++r)
        alast[r] = (float)aB[(size_t)(j0 + r + 8 * hi) * DA + DQK];

    // ---- GEMM over i in chunks of 128 (8 waves x 16 columns) ----
    for (int chunk = 0; chunk < 16; ++chunk) {
        const int icol = chunk * 128 + wave * 16 + n16;    // global i of this lane's column
        const _Float16* prow = qkB + (size_t)icol * DQK;   // P row i (first 64 elems = qk)
        if (chunk < 15)
            __builtin_prefetch(prow + 128 * DQK, 0, 1);    // next chunk's B rows

        // B fragments: K x N = 32 x 16, lane = column N, lanes>=16 hold K-blocks {8..15, 24..31}
        const int kb = hi ? 8 : 0;
        v16h B0, B1;
        *(uint4*)&B0         = *(const uint4*)(prow + kb);
        *(((uint4*)&B0) + 1) = *(const uint4*)(prow + kb + 16);
        *(uint4*)&B1         = *(const uint4*)(prow + 32 + kb);
        *(((uint4*)&B1) + 1) = *(const uint4*)(prow + 32 + kb + 16);

        v8f acc = {};
        acc = __builtin_amdgcn_wmma_f32_16x16x32_f16(false, A0, false, B0,
                                                     (short)0, acc, false, false);
        acc = __builtin_amdgcn_wmma_f32_16x16x32_f16(false, A1, false, B1,
                                                     (short)0, acc, false, false);

        const float qcv = qcB[icol];
        const float qsv = qsB[icol];
#pragma unroll
        for (int r = 0; r < 8; ++r) {
            // D layout: VGPR r -> M = r + 8*hi (j-row), N = n16 (i-col)
            float sc = acc[r] + qcv * alast[r];       // + P[i,64]*a[j,64]
            float v  = qsv * sc;                      // result[j,i] = Qs[i]*scores[i,j]
            v = (v == v) ? v : 0.0f;                  // NaN -> 0
            strip[(r + 8 * hi) * SLEN + icol] = (_Float16)v;
        }
    }
    __syncthreads();

    // ---- top-32 per j-row (wave handles 2 rows), then write mask row ----
    const float NEG = -3.0e38f;
#pragma unroll 1
    for (int rr = 0; rr < 2; ++rr) {
        const int jr = wave * 2 + rr;                 // 0..15
        float vals[64];
#pragma unroll
        for (int k = 0; k < 64; ++k)
            vals[k] = (float)strip[jr * SLEN + lane + (k << 5)];

        int sel = 0;
#pragma unroll 1
        for (int t = 0; t < 32; ++t) {
            float m = NEG; int mi = 0;
#pragma unroll
            for (int k = 0; k < 64; ++k) {
                const int idx = lane + (k << 5);
                if (vals[k] > m) { m = vals[k]; mi = idx; }
            }
#pragma unroll
            for (int off = 16; off > 0; off >>= 1) {
                const float om  = __shfl_xor(m, off, 32);
                const int   omi = __shfl_xor(mi, off, 32);
                if (om > m || (om == m && omi < mi)) { m = om; mi = omi; }
            }
            // invalidate the winner (constant-indexed, stays in VGPRs)
#pragma unroll
            for (int k = 0; k < 64; ++k)
                if (lane + (k << 5) == mi) vals[k] = NEG;
            if (t == lane) sel = mi;                  // lane t records rank-t index
        }

        float* orow = out + ((size_t)bh * SLEN + (j0 + jr)) * SLEN;
#pragma unroll
        for (int k = 0; k < 64; ++k)
            orow[lane + (k << 5)] = -10000.0f;
        orow[sel] = 0.0f;    // same-wave stores are ordered after the memset stores
    }
}

// ---------------------------------------------------------------------------
extern "C" void kernel_launch(void* const* d_in, const int* in_sizes, int n_in,
                              void* d_out, int out_size, void* d_ws, size_t ws_size,
                              hipStream_t stream) {
    const _Float16* qk = (const _Float16*)d_in[0];
    const _Float16* a  = (const _Float16*)d_in[1];
    float* out = (float*)d_out;

    const int rows = in_sizes[0] / DQK;     // B*H*S = 32768
    const int nBH  = rows / SLEN;           // 16

    float*        norm2    = (float*)d_ws;                 // rows f32
    unsigned int* phi_bits = (unsigned int*)norm2 + rows;  // 1 u32 (+pad)
    float*        qc       = (float*)norm2 + rows + 64;    // rows f32
    float*        qs       = qc + rows;                    // rows f32

    hipLaunchKernelGGL(init_phi, dim3(1), dim3(1), 0, stream, phi_bits);
    hipLaunchKernelGGL(norms_phi, dim3(rows / 256), dim3(256), 0, stream,
                       qk, norm2, phi_bits);
    hipLaunchKernelGGL(qs_qc, dim3(rows / 256), dim3(256), 0, stream,
                       qk, a, norm2, phi_bits, qc, qs);
    hipLaunchKernelGGL(gemm_topk, dim3(nBH * (SLEN / 16)), dim3(256), 0, stream,
                       qk, a, qc, qs, out);
}
